// RNN_24146306138158
// MI455X (gfx1250) — compile-verified, matrix operand order checked by reading, not run on hardware
//
#include <hip/hip_runtime.h>
#include <hip/hip_bf16.h>

typedef __attribute__((ext_vector_type(16))) _Float16 v16h;
typedef __attribute__((ext_vector_type(8)))  float    v8f;

#define B_   2048
#define T_   128
#define I_   130
#define H_   10
#define G3   30     // 3*H (gates r,z,n)
#define KP   160    // K padded to 5 tiles of 32
#define NP   32     // 3H padded to 2 tiles of 16
#define TC   8      // timesteps per chunk == waves per block
#define ROWS 16     // batch rows per workgroup
#define D1_  20

// ---------------------------------------------------------------------------
// Fused GRU: each workgroup owns 16 batch rows of ONE sequence (gru selected
// by blockIdx). Input projection uses v_wmma_f32_16x16x32_f16 (f32 accum);
// all weight fragments are hoisted into VGPRs before the time loop; the
// sequential recurrence runs fully inside LDS/registers.
// ---------------------------------------------------------------------------
__global__ __launch_bounds__(256) void gru_fused_kernel(
    const float* __restrict__ ru,      const float* __restrict__ en,
    const float* __restrict__ ruWih,   const float* __restrict__ ruWhh,
    const float* __restrict__ ruBih,   const float* __restrict__ ruBhh,
    const float* __restrict__ enWih,   const float* __restrict__ enWhh,
    const float* __restrict__ enBih,   const float* __restrict__ enBhh,
    float* __restrict__ hws)           // [2][B][H] last hidden states
{
    const int gru  = blockIdx.x >> 7;        // 0 = ru, 1 = en
    const int tile = blockIdx.x & 127;       // batch tile
    const float* __restrict__ x    = gru ? en    : ru;
    const float* __restrict__ Wih  = gru ? enWih : ruWih;
    const float* __restrict__ Whh  = gru ? enWhh : ruWhh;
    const float* __restrict__ bih  = gru ? enBih : ruBih;
    const float* __restrict__ bhh  = gru ? enBhh : ruBhh;

    __shared__ _Float16 sW[KP][NP];          // Wih^T padded, f16 (10 KB)
    __shared__ float    sBih[NP];
    __shared__ float    sWhh[G3][H_];
    __shared__ float    sBhh[G3];
    __shared__ float    sXi[TC][ROWS][NP];   // projected inputs per chunk (16 KB)
    __shared__ float    sH[2][ROWS][H_];     // double-buffered hidden state

    const int tid = threadIdx.x;

    // ---- stage weights into LDS (padded / converted) ----
    for (int idx = tid; idx < KP * NP; idx += 256) {
        int k = idx / NP, n = idx % NP;
        float w = (k < I_ && n < G3) ? Wih[n * I_ + k] : 0.0f;
        sW[k][n] = (_Float16)w;
    }
    if (tid < NP)        sBih[tid] = (tid < G3) ? bih[tid] : 0.0f;
    for (int idx = tid; idx < G3 * H_; idx += 256)
        sWhh[idx / H_][idx % H_] = Whh[idx];
    if (tid < G3)        sBhh[tid] = bhh[tid];
    if (tid < ROWS * H_) sH[0][tid / H_][tid % H_] = 0.0f;
    __syncthreads();

    const int wave  = tid >> 5;
    const int lane  = tid & 31;
    const int m     = lane & 15;             // A row / C column / B column
    const int khalf = (lane >> 4) * 8;       // K sub-offset per ISA A/B layout
    const int b0    = tile * ROWS;

    // ---- hoist the 10 B-matrix fragments into VGPRs (loop-invariant) ----
    v16h Bf[5][2];
#pragma unroll
    for (int kt = 0; kt < 5; ++kt) {
        const int kb = kt * 32 + khalf;
#pragma unroll
        for (int nt = 0; nt < 2; ++nt) {
            v16h f;
#pragma unroll
            for (int i = 0; i < 8; ++i) {
                f[i]     = sW[kb + i][nt * 16 + m];
                f[i + 8] = sW[kb + 16 + i][nt * 16 + m];
            }
            Bf[kt][nt] = f;
        }
    }
    const float bias0 = sBih[m];
    const float bias1 = sBih[16 + m];

    // ---- hoist this thread's recurrent weights into registers ----
    float Wr[H_], Wz[H_], Wn[H_], br = 0.f, bz = 0.f, bn = 0.f;
    const int rrow = tid / H_;               // batch row for recurrence
    const int rj   = tid % H_;               // hidden index for recurrence
    if (tid < ROWS * H_) {
#pragma unroll
        for (int k = 0; k < H_; ++k) {
            Wr[k] = sWhh[rj][k];
            Wz[k] = sWhh[H_ + rj][k];
            Wn[k] = sWhh[2 * H_ + rj][k];
        }
        br = sBhh[rj];
        bz = sBhh[H_ + rj];
        bn = sBhh[2 * H_ + rj];
    }

    // base pointer for this lane's A-matrix row, this wave's timestep stream
    const float* __restrict__ xrow =
        x + ((size_t)(b0 + m) * T_ + wave) * I_;

    int cur = 0;
    for (int t0 = 0; t0 < T_; t0 += TC) {
        // ================= input-projection GEMM (one timestep per wave) ===
        const float* __restrict__ xp = xrow + (size_t)t0 * I_;
        if (t0 + TC < T_)
            __builtin_prefetch(xp + (size_t)TC * I_, 0, 1); // next chunk

        v8f acc0 = {}; v8f acc1 = {};
#pragma unroll
        for (int kt = 0; kt < 5; ++kt) {
            const int kb = kt * 32 + khalf;
            v16h a;
#pragma unroll
            for (int i = 0; i < 8; ++i) {
                const int k0 = kb + i;
                const int k1 = kb + 16 + i;
                a[i]     = (k0 < I_) ? (_Float16)xp[k0] : (_Float16)0.0f;
                a[i + 8] = (k1 < I_) ? (_Float16)xp[k1] : (_Float16)0.0f;
            }
            acc0 = __builtin_amdgcn_wmma_f32_16x16x32_f16(
                false, a, false, Bf[kt][0], (short)0, acc0, false, false);
            acc1 = __builtin_amdgcn_wmma_f32_16x16x32_f16(
                false, a, false, Bf[kt][1], (short)0, acc1, false, false);
        }

        // C layout: vgpr v -> M = v + 8*(lane>=16), N = lane&15
#pragma unroll
        for (int v = 0; v < 8; ++v) {
            const int mm = v + khalf;
            sXi[wave][mm][m]      = acc0[v] + bias0;
            sXi[wave][mm][16 + m] = acc1[v] + bias1;
        }
        __syncthreads();

        // ================= sequential GRU recurrence over the chunk ========
        for (int s = 0; s < TC; ++s) {
            if (tid < ROWS * H_) {
                float gr = br, gz = bz, gn = bn;
#pragma unroll
                for (int k = 0; k < H_; ++k) {
                    const float hk = sH[cur][rrow][k];
                    gr += Wr[k] * hk;
                    gz += Wz[k] * hk;
                    gn += Wn[k] * hk;
                }
                const float xr = sXi[s][rrow][rj];
                const float xz = sXi[s][rrow][H_ + rj];
                const float xn = sXi[s][rrow][2 * H_ + rj];
                const float r  = 1.0f / (1.0f + __expf(-(xr + gr)));
                const float z  = 1.0f / (1.0f + __expf(-(xz + gz)));
                const float n  = tanhf(xn + r * gn);
                sH[cur ^ 1][rrow][rj] =
                    (1.0f - z) * n + z * sH[cur][rrow][rj];
            }
            cur ^= 1;
            __syncthreads();
        }
    }

    // ---- write last hidden state ----
    if (tid < ROWS * H_) {
        hws[(size_t)gru * B_ * H_ + (size_t)(b0 + rrow) * H_ + rj] =
            sH[cur][rrow][rj];
    }
}

// ---------------------------------------------------------------------------
// Head: out[b] = W2 @ (W1 @ concat(h_ru[b], h_en[b]) + b1) + b2
// ---------------------------------------------------------------------------
__global__ __launch_bounds__(256) void gru_head_kernel(
    const float* __restrict__ hws,   // [2][B][H]
    const float* __restrict__ W1,    // [20][20]
    const float* __restrict__ b1,    // [20]
    const float* __restrict__ W2,    // [1][20]
    const float* __restrict__ b2,    // [1]
    float* __restrict__ out)         // [B]
{
    const int b = blockIdx.x * blockDim.x + threadIdx.x;
    if (b >= B_) return;
    float c[2 * H_];
#pragma unroll
    for (int j = 0; j < H_; ++j) {
        c[j]      = hws[(size_t)b * H_ + j];
        c[H_ + j] = hws[(size_t)B_ * H_ + (size_t)b * H_ + j];
    }
    float acc = b2[0];
#pragma unroll
    for (int d = 0; d < D1_; ++d) {
        float h1 = b1[d];
#pragma unroll
        for (int k = 0; k < 2 * H_; ++k)
            h1 += W1[d * (2 * H_) + k] * c[k];
        acc += W2[d] * h1;
    }
    out[b] = acc;
}

extern "C" void kernel_launch(void* const* d_in, const int* in_sizes, int n_in,
                              void* d_out, int out_size, void* d_ws, size_t ws_size,
                              hipStream_t stream) {
    const float* ru     = (const float*)d_in[0];
    const float* en     = (const float*)d_in[1];
    const float* ruWih  = (const float*)d_in[2];
    const float* ruWhh  = (const float*)d_in[3];
    const float* ruBih  = (const float*)d_in[4];
    const float* ruBhh  = (const float*)d_in[5];
    const float* enWih  = (const float*)d_in[6];
    const float* enWhh  = (const float*)d_in[7];
    const float* enBih  = (const float*)d_in[8];
    const float* enBhh  = (const float*)d_in[9];
    const float* W1     = (const float*)d_in[10];
    const float* b1     = (const float*)d_in[11];
    const float* W2     = (const float*)d_in[12];
    const float* b2     = (const float*)d_in[13];

    float* hws = (float*)d_ws;                 // 2*B*H floats = 160 KB
    float* out = (float*)d_out;

    gru_fused_kernel<<<256, 256, 0, stream>>>(
        ru, en, ruWih, ruWhh, ruBih, ruBhh,
        enWih, enWhh, enBih, enBhh, hws);

    gru_head_kernel<<<(B_ + 255) / 256, 256, 0, stream>>>(
        hws, W1, b1, W2, b2, out);
}
